// Quantize_49426483642790
// MI455X (gfx1250) — matrix operand order repaired and checked by reading, not compiled, not run on hardware
//
#include <hip/hip_runtime.h>
#include <hip/hip_bf16.h>

// ---------------------------------------------------------------------------
// VQ-VAE quantize + EMA update for MI455X (gfx1250, wave32, WMMA).
// dist GEMM fused with argmin using v_wmma_f32_16x16x32_bf16 with a hi/lo
// bf16 split (error-compensated product). Codebook B-operands are staged in
// LDS per block with global_load_async_to_lds_b128 (ASYNCcnt path) and each
// wave processes 2 row-tiles per B load (register blocking).
// ---------------------------------------------------------------------------

#define N_ROWS   65536   // 64*32*32
#define D_DIM    64
#define K_CODES  512
#define DECAY    0.99f
#define ONE_M_D  0.01f
#define EPS_F    1e-5f

#define PACKB_BYTES 131072   // bh (64 KB) + bl (64 KB), contiguous in ws

typedef __attribute__((ext_vector_type(16))) __bf16 v16bf;
typedef __attribute__((ext_vector_type(8)))  float  v8f;
typedef __attribute__((ext_vector_type(4)))  float  v4f;

__device__ __forceinline__ __bf16 f32_to_bf16_trunc(float x) {
    union { float f; unsigned u; } a; a.f = x;
    union { unsigned short s; __bf16 b; } r;
    r.s = (unsigned short)(a.u >> 16);
    return r.b;
}
__device__ __forceinline__ float bf16_to_f32(__bf16 b) {
    union { unsigned short s; __bf16 b; } r; r.b = b;
    union { float f; unsigned u; } a;
    a.u = ((unsigned)r.s) << 16;
    return a.f;
}

// ---------------------------------------------------------------------------
// Kernel 1: zero accumulators, compute ||e||^2 per code, and pre-swizzle the
// codebook into the per-lane WMMA B-operand layout (hi and lo bf16 planes).
// Linear packed index = ((tile*2 + chunk)*32 + lane)*16 + j
//   column n = tile*16 + (lane&15); d = chunk*32 + (lane>>4)*16 + j
// ---------------------------------------------------------------------------
__global__ __launch_bounds__(256) void vq_pack(
    const float* __restrict__ embed,   // [64,512]
    __bf16* __restrict__ bh, __bf16* __restrict__ bl,
    float* __restrict__ esq,           // [512]
    float* __restrict__ count,         // [512]
    float* __restrict__ embed_sum,     // [64*512]
    float* __restrict__ sse)
{
    const int tid = blockIdx.x * 256 + threadIdx.x;   // 32768 threads total

    // zero the accumulators (deterministic per-call re-init)
    embed_sum[tid] = 0.0f;
    if (tid < K_CODES) {
        count[tid] = 0.0f;
        float s = 0.0f;
        #pragma unroll 8
        for (int d = 0; d < D_DIM; ++d) {
            float e = embed[d * K_CODES + tid];
            s += e * e;
        }
        esq[tid] = s;
    }
    if (tid == 0) *sse = 0.0f;

    // pack B operand (hi/lo split)
    const int j    = tid & 15;
    const int lane = (tid >> 4) & 31;
    const int c    = (tid >> 9) & 1;
    const int t    = tid >> 10;
    const int n    = t * 16 + (lane & 15);
    const int d    = c * 32 + (lane >> 4) * 16 + j;
    const float e  = embed[d * K_CODES + n];
    const __bf16 h = f32_to_bf16_trunc(e);
    const __bf16 l = f32_to_bf16_trunc(e - bf16_to_f32(h));
    bh[tid] = h;
    bl[tid] = l;
}

// ---------------------------------------------------------------------------
// Kernel 2: fused dist-GEMM + argmin + quantize gather + scatter stats.
// Block = 256 threads = 8 waves; wave w owns 32 rows (2 row-tiles), block
// owns 256 rows. Packed codebook staged in LDS (dynamic smem, 128 KB) via
// async global->LDS copies.
// ---------------------------------------------------------------------------
__global__ __launch_bounds__(256) void vq_main(
    const float* __restrict__ x,        // [65536,64]
    const float* __restrict__ embed,    // [64,512]
    const __bf16* __restrict__ bh,      // packed (bl contiguous after it)
    const float* __restrict__ esq,      // [512]
    float* __restrict__ out_q,          // [65536,64]
    float* __restrict__ out_ind,        // [65536] (indices as float)
    float* __restrict__ count,
    float* __restrict__ embed_sum,
    float* __restrict__ sse)
{
    extern __shared__ char s_dyn[];                 // 128 KB: packed bh|bl
    __shared__ float s_minv[256 * 16];
    __shared__ int   s_mini[256 * 16];
    __shared__ int   s_idx[256];
    __shared__ float s_red[256];

    const int tid  = threadIdx.x;
    const int wave = tid >> 5;
    const int lane = tid & 31;
    const int half = lane >> 4;        // 0 or 1
    const int lx   = lane & 15;
    const int blockRow = blockIdx.x * 256;
    const int rowBase  = blockRow + wave * 32;

    // ---- Stage packed codebook into LDS with async global->LDS b128 copies.
    // 131072 B / 16 B = 8192 units = 32 per thread. ASYNCcnt-tracked.
    {
        const char* src = (const char*)bh;
        #pragma unroll 4
        for (int i = 0; i < 32; ++i) {
            const int unit = i * 256 + tid;
            const unsigned lds_off =
                (unsigned)(uintptr_t)(s_dyn + (size_t)unit * 16);
            const unsigned long long gaddr =
                (unsigned long long)(uintptr_t)(src + (size_t)unit * 16);
            asm volatile("global_load_async_to_lds_b128 %0, %1, off"
                         :: "v"(lds_off), "v"(gaddr) : "memory");
        }
        asm volatile("s_wait_asynccnt 0x0" ::: "memory");
    }

    const __bf16* s_bh = (const __bf16*)(s_dyn);
    const __bf16* s_bl = (const __bf16*)(s_dyn + 65536);

    // ---- Load & convert A operands for 2 row-tiles into hi/lo bf16.
    // A k order used here (contiguous, same as packed B): lanes 0-15 row m = lx (K 0-15); lanes 16-31 same
    // rows, K 16-31. Per lane/chunk: one contiguous 16-float run.
    v16bf ah[2][2], al[2][2];          // [row-tile][chunk]
    #pragma unroll
    for (int rt = 0; rt < 2; ++rt) {
        const int R = rowBase + rt * 16 + lx;
        #pragma unroll
        for (int c = 0; c < 2; ++c) {
            const float* p = x + (size_t)R * D_DIM + c * 32 + half * 16;
            v4f q0 = *(const v4f*)(p);
            v4f q1 = *(const v4f*)(p + 4);
            v4f q2 = *(const v4f*)(p + 8);
            v4f q3 = *(const v4f*)(p + 12);
            float vals[16];
            #pragma unroll
            for (int j = 0; j < 4; ++j) {
                vals[j]      = q0[j];
                vals[4 + j]  = q1[j];
                vals[8 + j]  = q2[j];
                vals[12 + j] = q3[j];
            }
            #pragma unroll
            for (int j = 0; j < 16; ++j) {
                __bf16 h = f32_to_bf16_trunc(vals[j]);
                ah[rt][c][j] = h;
                al[rt][c][j] = f32_to_bf16_trunc(vals[j] - bf16_to_f32(h));
            }
        }
    }

    __syncthreads();   // LDS codebook visible to all waves

    // ---- Running min over the 32 column tiles (each 16 codes).
    float minv[2][8];
    int   mini[2][8];
    #pragma unroll
    for (int rt = 0; rt < 2; ++rt)
        #pragma unroll
        for (int i = 0; i < 8; ++i) { minv[rt][i] = 3.4e38f; mini[rt][i] = 0; }

    for (int t = 0; t < 32; ++t) {
        const int n  = t * 16 + lx;             // this lane's code column
        const float eq = esq[n];

        const v16bf b0h = *(const v16bf*)(s_bh + ((t * 2 + 0) * 32 + lane) * 16);
        const v16bf b0l = *(const v16bf*)(s_bl + ((t * 2 + 0) * 32 + lane) * 16);
        const v16bf b1h = *(const v16bf*)(s_bh + ((t * 2 + 1) * 32 + lane) * 16);
        const v16bf b1l = *(const v16bf*)(s_bl + ((t * 2 + 1) * 32 + lane) * 16);

        v8f acc0 = {}, acc1 = {};
        acc0 = __builtin_amdgcn_wmma_f32_16x16x32_bf16(false, ah[0][0], false, b0h, (short)0, acc0, false, false);
        acc1 = __builtin_amdgcn_wmma_f32_16x16x32_bf16(false, ah[1][0], false, b0h, (short)0, acc1, false, false);
        acc0 = __builtin_amdgcn_wmma_f32_16x16x32_bf16(false, al[0][0], false, b0h, (short)0, acc0, false, false);
        acc1 = __builtin_amdgcn_wmma_f32_16x16x32_bf16(false, al[1][0], false, b0h, (short)0, acc1, false, false);
        acc0 = __builtin_amdgcn_wmma_f32_16x16x32_bf16(false, ah[0][0], false, b0l, (short)0, acc0, false, false);
        acc1 = __builtin_amdgcn_wmma_f32_16x16x32_bf16(false, ah[1][0], false, b0l, (short)0, acc1, false, false);
        acc0 = __builtin_amdgcn_wmma_f32_16x16x32_bf16(false, ah[0][1], false, b1h, (short)0, acc0, false, false);
        acc1 = __builtin_amdgcn_wmma_f32_16x16x32_bf16(false, ah[1][1], false, b1h, (short)0, acc1, false, false);
        acc0 = __builtin_amdgcn_wmma_f32_16x16x32_bf16(false, al[0][1], false, b1h, (short)0, acc0, false, false);
        acc1 = __builtin_amdgcn_wmma_f32_16x16x32_bf16(false, al[1][1], false, b1h, (short)0, acc1, false, false);
        acc0 = __builtin_amdgcn_wmma_f32_16x16x32_bf16(false, ah[0][1], false, b1l, (short)0, acc0, false, false);
        acc1 = __builtin_amdgcn_wmma_f32_16x16x32_bf16(false, ah[1][1], false, b1l, (short)0, acc1, false, false);

        // dist (minus the per-row constant ||x||^2) = ||e||^2 - 2 x.e
        #pragma unroll
        for (int i = 0; i < 8; ++i) {
            float dv0 = __builtin_fmaf(-2.0f, acc0[i], eq);
            bool b0 = dv0 < minv[0][i];
            minv[0][i] = b0 ? dv0 : minv[0][i];
            mini[0][i] = b0 ? n   : mini[0][i];
            float dv1 = __builtin_fmaf(-2.0f, acc1[i], eq);
            bool b1 = dv1 < minv[1][i];
            minv[1][i] = b1 ? dv1 : minv[1][i];
            mini[1][i] = b1 ? n   : mini[1][i];
        }
    }

    // ---- Cross-lane argmin: C layout puts row m = i + 8*half in VGPR i,
    // column class = lx. Reduce the 16 classes per row through LDS.
    #pragma unroll
    for (int rt = 0; rt < 2; ++rt) {
        #pragma unroll
        for (int i = 0; i < 8; ++i) {
            const int r = wave * 32 + rt * 16 + i + half * 8;
            s_minv[r * 16 + lx] = minv[rt][i];
            s_mini[r * 16 + lx] = mini[rt][i];
        }
    }
    __syncthreads();

    {
        const float* pv = &s_minv[tid * 16];
        const int*   pi = &s_mini[tid * 16];
        float bv = pv[0]; int bi = pi[0];
        #pragma unroll
        for (int c = 1; c < 16; ++c) {
            if (pv[c] < bv) { bv = pv[c]; bi = pi[c]; }
        }
        s_idx[tid] = bi;
        out_ind[blockRow + tid] = (float)bi;
        atomicAdd(&count[bi], 1.0f);
    }
    __syncthreads();

    // ---- Coalesced quantize gather/store + SSE + embed_sum scatter.
    float part = 0.0f;
    for (int e = tid; e < 256 * D_DIM; e += 256) {
        const int r  = e >> 6;
        const int d  = e & 63;
        const int gi = s_idx[r];
        const size_t grow = (size_t)(blockRow + r);
        const float q  = embed[d * K_CODES + gi];
        const float xv = x[grow * D_DIM + d];
        out_q[grow * D_DIM + d] = q;
        const float df = q - xv;
        part += df * df;
        atomicAdd(&embed_sum[d * K_CODES + gi], xv);
    }
    s_red[tid] = part;
    __syncthreads();
    for (int s = 128; s > 0; s >>= 1) {
        if (tid < s) s_red[tid] += s_red[tid + s];
        __syncthreads();
    }
    if (tid == 0) atomicAdd(sse, s_red[0]);
}

// ---------------------------------------------------------------------------
// Kernel 3: EMA update + Laplace smoothing + diff. One block of 512 threads.
// ---------------------------------------------------------------------------
__global__ __launch_bounds__(512) void vq_final(
    const float* __restrict__ cluster_size,  // [512]
    const float* __restrict__ embed_avg,     // [64,512]
    const float* __restrict__ count,
    const float* __restrict__ embed_sum,
    const float* __restrict__ sse,
    float* __restrict__ out_diff,            // [1]
    float* __restrict__ out_ncs,             // [512]
    float* __restrict__ out_nea,             // [64,512]
    float* __restrict__ out_emb_new)         // [64,512]
{
    __shared__ float s_sum[512];
    const int t = threadIdx.x;

    const float ncs = cluster_size[t] * DECAY + ONE_M_D * count[t];
    out_ncs[t] = ncs;
    s_sum[t] = ncs;
    __syncthreads();
    for (int s = 256; s > 0; s >>= 1) {
        if (t < s) s_sum[t] += s_sum[t + s];
        __syncthreads();
    }
    const float n  = s_sum[0];
    const float cs = (ncs + EPS_F) / (n + (float)K_CODES * EPS_F) * n;
    const float inv = 1.0f / cs;

    #pragma unroll 8
    for (int d = 0; d < D_DIM; ++d) {
        const int idx = d * K_CODES + t;
        const float nea = embed_avg[idx] * DECAY + ONE_M_D * embed_sum[idx];
        out_nea[idx] = nea;
        out_emb_new[idx] = nea * inv;
    }
    if (t == 0) *out_diff = (*sse) * (1.0f / (float)((size_t)N_ROWS * D_DIM));
}

// ---------------------------------------------------------------------------
extern "C" void kernel_launch(void* const* d_in, const int* in_sizes, int n_in,
                              void* d_out, int out_size, void* d_ws, size_t ws_size,
                              hipStream_t stream) {
    const float* x            = (const float*)d_in[0];   // [64,32,32,64]
    const float* embed        = (const float*)d_in[1];   // [64,512]
    const float* cluster_size = (const float*)d_in[2];   // [512]
    const float* embed_avg    = (const float*)d_in[3];   // [64,512]

    // d_out layout (flat, return order):
    float* out_q       = (float*)d_out;                        // 4194304
    float* out_diff    = out_q + 4194304;                      // 1
    float* out_ind     = out_diff + 1;                         // 65536
    float* out_emb_new = out_ind + 65536;                      // 32768
    float* out_ncs     = out_emb_new + 32768;                  // 512
    float* out_nea     = out_ncs + 512;                        // 32768

    // workspace layout (bh and bl MUST stay contiguous: staged as one copy)
    char* ws = (char*)d_ws;
    __bf16* bh        = (__bf16*)(ws + 0);          //  65536 B
    __bf16* bl        = (__bf16*)(ws + 65536);      //  65536 B
    float*  esq       = (float*)(ws + 131072);      //   2048 B
    float*  count     = (float*)(ws + 133120);      //   2048 B
    float*  embed_sum = (float*)(ws + 135168);      // 131072 B
    float*  sse       = (float*)(ws + 266240);      //      4 B

    vq_pack<<<128, 256, 0, stream>>>(embed, bh, bl, esq, count, embed_sum, sse);
    vq_main<<<256, 256, PACKB_BYTES, stream>>>(x, embed, bh, esq,
                                               out_q, out_ind, count, embed_sum, sse);
    vq_final<<<1, 512, 0, stream>>>(cluster_size, embed_avg, count, embed_sum, sse,
                                    out_diff, out_ncs, out_nea, out_emb_new);
}
